// EfficientGNNClassifier_57071525429489
// MI455X (gfx1250) — matrix operand order repaired and checked by reading, not compiled, not run on hardware
//
#include <hip/hip_runtime.h>
#include <hip/hip_bf16.h>

// ---------------------------------------------------------------------------
// Types for CDNA5 WMMA (wave32): V_WMMA_F32_16X16X32_BF16
// ---------------------------------------------------------------------------
typedef __attribute__((ext_vector_type(16))) __bf16 v16bf;
typedef __attribute__((ext_vector_type(8)))  __bf16 v8bf;
typedef __attribute__((ext_vector_type(8)))  float  v8f;

static __device__ inline __bf16 f32_to_bf16_rne(float f) {
    union { float f; unsigned int u; } cv; cv.f = f;
    unsigned int u = cv.u;
    u += 0x7FFFu + ((u >> 16) & 1u);           // round-to-nearest-even
    unsigned short h = (unsigned short)(u >> 16);
    union { unsigned short s; __bf16 b; } cb; cb.s = h;
    return cb.b;
}

static __device__ inline float bf16_to_f32(__bf16 b) {
    union { unsigned short s; __bf16 b; } cb; cb.b = b;
    union { unsigned int u; float f; } cv; cv.u = (unsigned int)cb.s << 16;
    return cv.f;
}

// ---------------------------------------------------------------------------
// Fragment loaders (CDNA5 ISA 7.12.2), operating on pre-converted
// bf16 data so the hot loop is pure b128 vector loads + WMMA.
// A 16x32 (MxK): lanes 0-15 -> M=lane, elems 0..7 = K+0..7, 8..15 = K+16..23;
//                lanes 16-31 -> same M, +8 K offset.
// ---------------------------------------------------------------------------
static __device__ inline v16bf load_a_frag(const __bf16* __restrict__ A,
                                           int row, int k0, int K, int lane) {
    const __bf16* p = A + (size_t)row * K + k0 + ((lane & 16) ? 8 : 0);
    v8bf lo = *(const v8bf*)(p);        // 16B aligned
    v8bf hi = *(const v8bf*)(p + 16);
    return __builtin_shufflevector(lo, hi, 0,1,2,3,4,5,6,7,8,9,10,11,12,13,14,15);
}

// B 32x16 (KxN) pre-packed "fragment-ready": per (tileN, kblock), per lane,
// 16 contiguous bf16 (=32B, one v16bf load).
static __device__ inline v16bf load_b_frag(const __bf16* __restrict__ Wp,
                                           int tileN, int kb, int KB, int lane) {
    const __bf16* p = Wp + ((size_t)((size_t)tileN * KB + kb) * 32 + lane) * 16;
    return *(const v16bf*)p;            // 32B aligned
}

// ---------------------------------------------------------------------------
// Fused dual-GEMM: out = epilogue(A1@W1 + A2@W2); epilogue acc*scale[c]+shift[c]
// (+ReLU). One wave computes a 16x(16*NT) output tile, reusing the A fragment
// across NT WMMAs. K2=0 skips the second GEMM. out is bf16 or f32.
// ---------------------------------------------------------------------------
template<int NT>
__global__ __launch_bounds__(32)
void gemm2_wmma(const __bf16* __restrict__ A1, const __bf16* __restrict__ W1p,
                const __bf16* __restrict__ A2, const __bf16* __restrict__ W2p,
                const float* __restrict__ scale, const float* __restrict__ shift,
                void* __restrict__ outp,
                int K1, int K2, int Nc, int do_relu, int out_bf16) {
    const int lane   = threadIdx.x;
    const int tileM  = blockIdx.x;
    const int tileN0 = blockIdx.y * NT;
    const int row    = tileM * 16 + (lane & 15);

    v8f acc[NT] = {};

    const int KB1 = K1 >> 5;
    for (int kb = 0; kb < KB1; ++kb) {
        __builtin_prefetch(A1 + (size_t)row * K1 + (kb + 1) * 32, 0, 0);
        v16bf a = load_a_frag(A1, row, kb * 32, K1, lane);
#pragma unroll
        for (int t = 0; t < NT; ++t) {
            v16bf b = load_b_frag(W1p, tileN0 + t, kb, KB1, lane);
            acc[t] = __builtin_amdgcn_wmma_f32_16x16x32_bf16(false, a, false, b,
                                                             (short)0, acc[t], false, false);
        }
    }
    const int KB2 = K2 >> 5;
    for (int kb = 0; kb < KB2; ++kb) {
        __builtin_prefetch(A2 + (size_t)row * K2 + (kb + 1) * 32, 0, 0);
        v16bf a = load_a_frag(A2, row, kb * 32, K2, lane);
#pragma unroll
        for (int t = 0; t < NT; ++t) {
            v16bf b = load_b_frag(W2p, tileN0 + t, kb, KB2, lane);
            acc[t] = __builtin_amdgcn_wmma_f32_16x16x32_bf16(false, a, false, b,
                                                             (short)0, acc[t], false, false);
        }
    }

    // C/D layout: lanes 0-15 -> rows +0..7, lanes 16-31 -> rows +8..15.
    const int rbase = tileM * 16 + ((lane & 16) ? 8 : 0);
#pragma unroll
    for (int t = 0; t < NT; ++t) {
        const int col = (tileN0 + t) * 16 + (lane & 15);
        const float sc = scale[col];
        const float sh = shift[col];
#pragma unroll
        for (int r = 0; r < 8; ++r) {
            float v = acc[t][r] * sc + sh;
            if (do_relu) v = fmaxf(v, 0.0f);
            if (out_bf16)
                ((__bf16*)outp)[(size_t)(rbase + r) * Nc + col] = f32_to_bf16_rne(v);
            else
                ((float*)outp)[(size_t)(rbase + r) * Nc + col] = v;
        }
    }
}

// ---------------------------------------------------------------------------
// Data-movement / epilogue helper kernels
// ---------------------------------------------------------------------------
__global__ void zero_f32(float* __restrict__ p, long long n) {
    long long i = (long long)blockIdx.x * blockDim.x + threadIdx.x;
    long long stride = (long long)gridDim.x * blockDim.x;
    for (; i < n; i += stride) p[i] = 0.0f;
}

__global__ void cvt_f32_to_bf16(const float* __restrict__ in, __bf16* __restrict__ out,
                                long long n) {
    long long i = (long long)blockIdx.x * blockDim.x + threadIdx.x;
    long long stride = (long long)gridDim.x * blockDim.x;
    for (; i < n; i += stride) out[i] = f32_to_bf16_rne(in[i]);
}

// Pack W[K,Nc] f32 into fragment-ready bf16:
//   Wp[((t*KB + kb)*32 + lane)*16 + e] = W[(kb*32 + (lane>>4)*8 + (e&7) + (e>>3)*16)*Nc + t*16 + (lane&15)]
__global__ void pack_w_bf16(const float* __restrict__ W, __bf16* __restrict__ Wp,
                            int K, int Nc) {
    long long idx = (long long)blockIdx.x * blockDim.x + threadIdx.x;
    long long tot = (long long)K * Nc;
    if (idx >= tot) return;
    const int KB = K >> 5;
    int e  = (int)(idx & 15);
    int l  = (int)((idx >> 4) & 31);
    long long blk = idx >> 9;
    int t  = (int)(blk / KB);
    int kb = (int)(blk % KB);
    int k  = kb * 32 + ((l >> 4) * 8) + (e & 7) + ((e >> 3) * 16);
    int n  = t * 16 + (l & 15);
    Wp[idx] = f32_to_bf16_rne(W[(size_t)k * Nc + n]);
}

__global__ void degree_kernel(const int* __restrict__ dst, float* __restrict__ deg, int E) {
    int e = blockIdx.x * blockDim.x + threadIdx.x;
    if (e < E) atomicAdd(&deg[dst[e]], 1.0f);
}

// One block per edge, one thread per feature (blockDim == D). Coalesced.
__global__ void scatter_add_bf16(const __bf16* __restrict__ x,
                                 const int* __restrict__ src, const int* __restrict__ dst,
                                 float* __restrict__ agg, int D) {
    int e = blockIdx.x;
    int s = src[e], d = dst[e];
    int f = threadIdx.x;
    atomicAdd(&agg[(size_t)d * D + f], bf16_to_f32(x[(size_t)s * D + f]));
}

__global__ void mean_div_bf16(const float* __restrict__ agg, const float* __restrict__ deg,
                              __bf16* __restrict__ out, int D) {
    int i = blockIdx.x;
    int f = threadIdx.x;
    out[(size_t)i * D + f] = f32_to_bf16_rne(agg[(size_t)i * D + f] / fmaxf(deg[i], 1.0f));
}

// Fold bias + eval BN into per-column scale/shift: acc*s + t
__global__ void bn_coeffs(const float* __restrict__ g, const float* __restrict__ beta,
                          const float* __restrict__ m, const float* __restrict__ v,
                          const float* __restrict__ bias,
                          float* __restrict__ s, float* __restrict__ t, int n) {
    int i = blockIdx.x * blockDim.x + threadIdx.x;
    if (i < n) {
        float sc = g[i] * rsqrtf(v[i] + 1e-5f);
        s[i] = sc;
        t[i] = beta[i] + (bias[i] - m[i]) * sc;
    }
}

__global__ void id_coeffs(const float* __restrict__ bias,
                          float* __restrict__ s, float* __restrict__ t, int n) {
    int i = blockIdx.x * blockDim.x + threadIdx.x;
    if (i < n) { s[i] = 1.0f; t[i] = bias[i]; }
}

__global__ void log_softmax16(float* __restrict__ out, int n) {
    int i = blockIdx.x * blockDim.x + threadIdx.x;
    if (i >= n) return;
    float* row = out + (size_t)i * 16;
    float mx = row[0];
#pragma unroll
    for (int c = 1; c < 16; ++c) mx = fmaxf(mx, row[c]);
    float sum = 0.0f;
#pragma unroll
    for (int c = 0; c < 16; ++c) sum += __expf(row[c] - mx);
    float lse = __logf(sum) + mx;
#pragma unroll
    for (int c = 0; c < 16; ++c) row[c] -= lse;
}

// ---------------------------------------------------------------------------
// Orchestration
// ---------------------------------------------------------------------------
extern "C" void kernel_launch(void* const* d_in, const int* in_sizes, int n_in,
                              void* d_out, int out_size, void* d_ws, size_t ws_size,
                              hipStream_t stream) {
    const float* x   = (const float*)d_in[0];
    const int*   ei  = (const int*)  d_in[1];
    const float* Wl1 = (const float*)d_in[2];
    const float* bl1 = (const float*)d_in[3];
    const float* Wr1 = (const float*)d_in[4];
    const float* g1  = (const float*)d_in[5];
    const float* b1  = (const float*)d_in[6];
    const float* m1  = (const float*)d_in[7];
    const float* v1  = (const float*)d_in[8];
    const float* Wl2 = (const float*)d_in[9];
    const float* bl2 = (const float*)d_in[10];
    const float* Wr2 = (const float*)d_in[11];
    const float* g2  = (const float*)d_in[12];
    const float* b2  = (const float*)d_in[13];
    const float* m2  = (const float*)d_in[14];
    const float* v2  = (const float*)d_in[15];
    const float* Wl3 = (const float*)d_in[16];
    const float* bl3 = (const float*)d_in[17];
    const float* Wr3 = (const float*)d_in[18];
    const float* Wf1 = (const float*)d_in[19];
    const float* bf1 = (const float*)d_in[20];
    const float* Wf2 = (const float*)d_in[21];
    const float* bf2 = (const float*)d_in[22];

    const int N = in_sizes[0] / 96;     // 50000
    const int E = in_sizes[1] / 2;      // 800000
    const int* src = ei;
    const int* dst = ei + E;

    // ---- workspace partition (256B-aligned regions) ----
    char* base = (char*)d_ws;
    size_t off = 0;
    auto alloc = [&](size_t bytes) -> void* {
        void* p = base + off;
        off = (off + bytes + 255) & ~(size_t)255;
        return p;
    };
    float*  deg   = (float*) alloc((size_t)N * 4);
    float*  agg   = (float*) alloc((size_t)N * 256 * 4);
    __bf16* meanb = (__bf16*)alloc((size_t)N * 256 * 2);
    __bf16* xb    = (__bf16*)alloc((size_t)N * 96  * 2);
    __bf16* h1    = (__bf16*)alloc((size_t)N * 256 * 2);
    __bf16* h2    = (__bf16*)alloc((size_t)N * 256 * 2);
    float*  scl   = (float*) alloc(256 * 4);
    float*  shf   = (float*) alloc(256 * 4);
    __bf16* wl1p  = (__bf16*)alloc((size_t)96  * 256 * 2);
    __bf16* wr1p  = (__bf16*)alloc((size_t)96  * 256 * 2);
    __bf16* wl2p  = (__bf16*)alloc((size_t)256 * 256 * 2);
    __bf16* wr2p  = (__bf16*)alloc((size_t)256 * 256 * 2);
    __bf16* wl3p  = (__bf16*)alloc((size_t)256 * 128 * 2);
    __bf16* wr3p  = (__bf16*)alloc((size_t)256 * 128 * 2);
    __bf16* wf1p  = (__bf16*)alloc((size_t)128 * 64  * 2);
    __bf16* wf2p  = (__bf16*)alloc((size_t)64  * 16  * 2);
    __bf16* h3 = h1;                      // [N,128], h1 dead after layer 2
    __bf16* h4 = h2;                      // [N,64],  h2 dead after fc1 input read? (dead after layer-3 GEMM)
    float* logits = (float*)d_out;        // [N,16]

    const int TM = N / 16;                // 3125 (exact)

    // ---- one-time per launch: pack weights, convert x, degrees ----
    auto packW = [&](const float* W, __bf16* Wp, int K, int Nc) {
        long long tot = (long long)K * Nc;
        pack_w_bf16<<<(int)((tot + 255) / 256), 256, 0, stream>>>(W, Wp, K, Nc);
    };
    packW(Wl1, wl1p, 96, 256);  packW(Wr1, wr1p, 96, 256);
    packW(Wl2, wl2p, 256, 256); packW(Wr2, wr2p, 256, 256);
    packW(Wl3, wl3p, 256, 128); packW(Wr3, wr3p, 256, 128);
    packW(Wf1, wf1p, 128, 64);  packW(Wf2, wf2p, 64, 16);

    cvt_f32_to_bf16<<<(int)(((long long)N * 96 + 255) / 256), 256, 0, stream>>>(
        x, xb, (long long)N * 96);

    zero_f32<<<(N + 255) / 256, 256, 0, stream>>>(deg, N);
    degree_kernel<<<(E + 255) / 256, 256, 0, stream>>>(dst, deg, E);

    // ---- Layer 1: h1 = relu(bn1(mean(x)@Wl1 + bl1 + x@Wr1)) ----
    zero_f32<<<(int)(((long long)N * 96 + 255) / 256), 256, 0, stream>>>(agg, (long long)N * 96);
    scatter_add_bf16<<<E, 96, 0, stream>>>(xb, src, dst, agg, 96);
    mean_div_bf16<<<N, 96, 0, stream>>>(agg, deg, meanb, 96);
    bn_coeffs<<<1, 256, 0, stream>>>(g1, b1, m1, v1, bl1, scl, shf, 256);
    gemm2_wmma<2><<<dim3(TM, 256 / 32), 32, 0, stream>>>(
        meanb, wl1p, xb, wr1p, scl, shf, h1, 96, 96, 256, 1, 1);

    // ---- Layer 2: h2 = relu(bn2(mean(h1)@Wl2 + bl2 + h1@Wr2)) ----
    zero_f32<<<(int)(((long long)N * 256 + 255) / 256), 256, 0, stream>>>(agg, (long long)N * 256);
    scatter_add_bf16<<<E, 256, 0, stream>>>(h1, src, dst, agg, 256);
    mean_div_bf16<<<N, 256, 0, stream>>>(agg, deg, meanb, 256);
    bn_coeffs<<<1, 256, 0, stream>>>(g2, b2, m2, v2, bl2, scl, shf, 256);
    gemm2_wmma<2><<<dim3(TM, 256 / 32), 32, 0, stream>>>(
        meanb, wl2p, h1, wr2p, scl, shf, h2, 256, 256, 256, 1, 1);

    // ---- Layer 3: h3 = relu(mean(h2)@Wl3 + bl3 + h2@Wr3) ----
    zero_f32<<<(int)(((long long)N * 256 + 255) / 256), 256, 0, stream>>>(agg, (long long)N * 256);
    scatter_add_bf16<<<E, 256, 0, stream>>>(h2, src, dst, agg, 256);
    mean_div_bf16<<<N, 256, 0, stream>>>(agg, deg, meanb, 256);
    id_coeffs<<<1, 128, 0, stream>>>(bl3, scl, shf, 128);
    gemm2_wmma<2><<<dim3(TM, 128 / 32), 32, 0, stream>>>(
        meanb, wl3p, h2, wr3p, scl, shf, h3, 256, 256, 128, 1, 1);

    // ---- FC1: h4 = relu(h3 @ Wf1 + bf1) ----
    id_coeffs<<<1, 64, 0, stream>>>(bf1, scl, shf, 64);
    gemm2_wmma<2><<<dim3(TM, 64 / 32), 32, 0, stream>>>(
        h3, wf1p, h3, wf1p, scl, shf, h4, 128, 0, 64, 1, 1);

    // ---- FC2: logits = h4 @ Wf2 + bf2 (f32 out) ----
    id_coeffs<<<1, 32, 0, stream>>>(bf2, scl, shf, 16);
    gemm2_wmma<1><<<dim3(TM, 1), 32, 0, stream>>>(
        h4, wf2p, h4, wf2p, scl, shf, logits, 64, 0, 16, 0, 0);

    // ---- log_softmax over 16 logits, in place on d_out ----
    log_softmax16<<<(N + 255) / 256, 256, 0, stream>>>(logits, N);
}
